// GlobalPointer_51307679318483
// MI455X (gfx1250) — matrix operand order repaired and checked by reading, not compiled, not run on hardware
//
#include <hip/hip_runtime.h>
#include <hip/hip_bf16.h>

typedef __attribute__((ext_vector_type(2)))  float    v2f;
typedef __attribute__((ext_vector_type(8)))  float    v8f;
typedef __attribute__((ext_vector_type(8)))  _Float16 v8h;
typedef __attribute__((ext_vector_type(16))) _Float16 v16h;

#define BATCH    8
#define NSEQ     1024
#define HIDDEN   1024
#define HEADS    12
#define HEAD_SZ  64
#define PROJC    (HEADS * HEAD_SZ * 2)   /* 1536 */

#define BM 128            /* block tile M */
#define BN 128            /* block tile N */
#define KT 16             /* K tile staged in LDS per step */
#define KSTEPS (HIDDEN / KT)
#define A_ELEMS (BM * KT) /* 2048 floats = 8KB */
#define B_ELEMS (KT * BN) /* 2048 floats = 8KB */

// -----------------------------------------------------------------------------
// Kernel 1: proj = x @ W + b, then RoPE, store q/k as f16 [b][h][token][d].
// Block 256 thr = 8 waves (4 M x 2 N), wave tile 32x64 -> 2x4 WMMA subtiles.
// K staged in double-buffered LDS via global_load_async_to_lds_b128 (ASYNCcnt),
// fragments fed from LDS to V_WMMA_F32_16X16X4_F32.
// Fragment layouts (wave32):
//   A (f32 16x4):  lane L: M=L%16, VGPR{0,1} = K = (L<16?0:2)+{0,1}
//   B (f32 4x16):  lane L: N=L%16, VGPR{0,1} = K = (L<16?0:2)+{0,1}
//   C (f32 16x16): lane L: N=L%16, VGPR v -> M = v + (L>=16)*8
// -----------------------------------------------------------------------------
__global__ __launch_bounds__(256) void proj_rope_k(
    const float* __restrict__ x, const float* __restrict__ W,
    const float* __restrict__ bias,
    _Float16* __restrict__ qws, _Float16* __restrict__ kws)
{
    __shared__ float lds[2][A_ELEMS + B_ELEMS];     // 32 KB double buffer

    const int tid  = threadIdx.x;
    const int lane = tid & 31;
    const int wave = tid >> 5;
    const int hgrp = lane >> 4;                     // half-wave select
    const int l16  = lane & 15;
    const int wm   = wave & 3;                      // wave M position (x32)
    const int wn   = wave >> 2;                     // wave N position (x64)
    const int mBase = blockIdx.y * BM;
    const int nBase = blockIdx.x * BN;

    // Issue async global->LDS copies for one K tile (2 A + 2 B b128 per thread).
    auto issue_tile = [&](int kb, int buf) {
        float* Ab = &lds[buf][0];
        float* Bb = &lds[buf][A_ELEMS];
        #pragma unroll
        for (int i = 0; i < 2; ++i) {
            const int id = tid * 2 + i;
            {   // A tile: BM x KT, row-major [row][k]
                const int row = id >> 2, c4 = id & 3;
                unsigned    la = (unsigned)(size_t)&Ab[row * KT + c4 * 4];
                const float* ga = x + (size_t)(mBase + row) * HIDDEN + kb + c4 * 4;
                asm volatile("global_load_async_to_lds_b128 %0, %1, off"
                             :: "v"(la), "v"(ga) : "memory");
            }
            {   // B tile: KT x BN, row-major [k][col]
                const int row = id >> 5, c4 = id & 31;
                unsigned    lb = (unsigned)(size_t)&Bb[row * BN + c4 * 4];
                const float* gb = W + (size_t)(kb + row) * PROJC + nBase + c4 * 4;
                asm volatile("global_load_async_to_lds_b128 %0, %1, off"
                             :: "v"(lb), "v"(gb) : "memory");
            }
        }
    };

    issue_tile(0, 0);
    asm volatile("s_wait_asynccnt 0x0" ::: "memory");
    __syncthreads();

    v8f acc[2][4] = {};
    for (int step = 0; step < KSTEPS; ++step) {
        const int buf = step & 1;
        if (step + 1 < KSTEPS) issue_tile((step + 1) * KT, buf ^ 1);

        const float* Ab = &lds[buf][0];
        const float* Bb = &lds[buf][A_ELEMS];
        #pragma unroll
        for (int kk = 0; kk < KT / 4; ++kk) {
            const int krow = kk * 4 + hgrp * 2;
            const v2f a0 = *(const v2f*)&Ab[(wm * 32 +      l16) * KT + krow];
            const v2f a1 = *(const v2f*)&Ab[(wm * 32 + 16 + l16) * KT + krow];
            #pragma unroll
            for (int ns = 0; ns < 4; ++ns) {
                v2f bf;
                bf.x = Bb[ krow      * BN + wn * 64 + ns * 16 + l16];
                bf.y = Bb[(krow + 1) * BN + wn * 64 + ns * 16 + l16];
                acc[0][ns] = __builtin_amdgcn_wmma_f32_16x16x4_f32(
                                 false, a0, false, bf, (short)0, acc[0][ns], false, false);
                acc[1][ns] = __builtin_amdgcn_wmma_f32_16x16x4_f32(
                                 false, a1, false, bf, (short)0, acc[1][ns], false, false);
            }
        }
        if (step + 1 < KSTEPS)
            asm volatile("s_wait_asynccnt 0x0" ::: "memory");
        __syncthreads();
    }

    // Epilogue: bias + RoPE + f16 store.
    #pragma unroll
    for (int ns = 0; ns < 4; ++ns) {
        const int colB = nBase + wn * 64 + ns * 16 + l16;   // fixed per lane
        const float bv = bias[colB];
        const int h = colB >> 7;
        const int r = colB & 127;
        _Float16* __restrict__ dst = (r & 64) ? kws : qws;
        const int d = r & 63;
        const float inv = powf(10000.f, (-2.f / (float)HEAD_SZ) * (float)(d >> 1));
        #pragma unroll
        for (int ms = 0; ms < 2; ++ms) {
            #pragma unroll
            for (int v = 0; v < 8; ++v) {
                const int mrow = mBase + wm * 32 + ms * 16 + v + hgrp * 8;
                const int bidx = mrow >> 10;
                const int pos  = mrow & (NSEQ - 1);
                const float t  = acc[ms][ns][v] + bv;
                // RoPE partner (feature d^1) is the adjacent lane, same VGPR.
                const float tp = __shfl_xor(t, 1, 32);
                float s, c;
                sincosf((float)pos * inv, &s, &c);
                const float rot = (d & 1) ? tp : -tp;
                dst[((((size_t)bidx * HEADS + h) * NSEQ) + pos) * HEAD_SZ + d] =
                    (_Float16)(t * c + rot * s);
            }
        }
    }
}

// -----------------------------------------------------------------------------
// Kernel 2: logits[b,h,m,n] = sum_d q[b,h,m,d]*k[b,h,n,d]; mask+causal+scale.
// One wave per 16(M) x 64(N) strip; V_WMMA_F32_16X16X32_F16, K=64 in 2 steps.
// A frag (f16 16x32, per ISA): lane<16 : V0-3 = K 0-7,  V4-7 = K 16-23
//                              lane>=16: V0-3 = K 8-15, V4-7 = K 24-31
// B frag (f16 32x16): lane L: N=L%16; same K order as A (K 0-7,16-23 / 8-15,24-31 per half)
// Output streamed with nontemporal stores (384 MB write-once).
// -----------------------------------------------------------------------------
__global__ __launch_bounds__(256) void logits_k(
    const _Float16* __restrict__ qws, const _Float16* __restrict__ kws,
    const int* __restrict__ mask, float* __restrict__ out)
{
    const int lane = threadIdx.x & 31;
    const int wave = threadIdx.x >> 5;
    const int hgrp = lane >> 4;
    const int l16  = lane & 15;
    const int bh = blockIdx.z;                 // 0..95
    const int b  = bh / HEADS;
    const int n0 = blockIdx.x * 64;            // N strip base
    const int mt = blockIdx.y * 8 + wave;      // 0..63
    const int m0 = mt * 16;

    const size_t base = (size_t)bh * NSEQ * HEAD_SZ;
    const _Float16* __restrict__ q = qws + base;
    const _Float16* __restrict__ k = kws + base;

    // A fragments for K chunks 0 and 1 (reused across 4 N sub-tiles).
    v16h afr[2];
    {
        const _Float16* ap = q + (size_t)(m0 + l16) * HEAD_SZ;
        #pragma unroll
        for (int kk = 0; kk < 2; ++kk) {
            const v8h lo = *(const v8h*)(ap + kk * 32 + hgrp * 8);
            const v8h hi = *(const v8h*)(ap + kk * 32 + 16 + hgrp * 8);
            v16h a;
            #pragma unroll
            for (int i = 0; i < 8; ++i) { a[i] = lo[i]; a[8 + i] = hi[i]; }
            afr[kk] = a;
        }
    }

    v8f acc[4] = {{}, {}, {}, {}};
    #pragma unroll
    for (int nt = 0; nt < 4; ++nt) {
        const _Float16* kp = k + (size_t)(n0 + nt * 16 + l16) * HEAD_SZ + hgrp * 8;
        #pragma unroll
        for (int kk = 0; kk < 2; ++kk) {
            const v8h b0 = *(const v8h*)(kp + kk * 32);
            const v8h b1 = *(const v8h*)(kp + kk * 32 + 16);
            v16h bf;
            #pragma unroll
            for (int i = 0; i < 8; ++i) { bf[i] = b0[i]; bf[8 + i] = b1[i]; }
            acc[nt] = __builtin_amdgcn_wmma_f32_16x16x32_f16(
                          false, afr[kk], false, bf, (short)0, acc[nt], false, false);
        }
    }

    // Epilogue mirrors reference order:
    //   where(mask_m & mask_n, L, -NEG); L -= (m<=n ? 0 : NEG); L /= 8
    const int* __restrict__ mrow = mask + b * NSEQ;
    float* __restrict__ ob = out + (size_t)bh * NSEQ * NSEQ;
    #pragma unroll
    for (int nt = 0; nt < 4; ++nt) {
        const int n  = n0 + nt * 16 + l16;
        const int mn = mrow[n];
        #pragma unroll
        for (int v = 0; v < 8; ++v) {
            const int m = m0 + v + hgrp * 8;
            float L = acc[nt][v];
            if (!(mrow[m] && mn)) L = -1.0e12f;
            if (m > n)            L -= 1.0e12f;
            __builtin_nontemporal_store(L * 0.125f, &ob[(size_t)m * NSEQ + n]);
        }
    }
}

extern "C" void kernel_launch(void* const* d_in, const int* in_sizes, int n_in,
                              void* d_out, int out_size, void* d_ws, size_t ws_size,
                              hipStream_t stream) {
    const float* x    = (const float*)d_in[0];
    const int*   mask = (const int*)d_in[1];
    const float* W    = (const float*)d_in[2];
    const float* bias = (const float*)d_in[3];
    float* out = (float*)d_out;

    _Float16* qws = (_Float16*)d_ws;                               // 12 MB
    _Float16* kws = qws + (size_t)BATCH * HEADS * NSEQ * HEAD_SZ;  // +12 MB

    // GEMM1 + RoPE: 128x128 block tiles -> grid (1536/128, 8192/128)
    proj_rope_k<<<dim3(PROJC / BN, (BATCH * NSEQ) / BM), 256, 0, stream>>>(
        x, W, bias, qws, kws);

    // GEMM2 + mask: (16 N-strips) x (64 M-tiles / 8 waves) x (96 batch*head)
    logits_k<<<dim3(NSEQ / 64, NSEQ / (16 * 8), BATCH * HEADS), 256, 0, stream>>>(
        qws, kws, mask, out);
}
